// ViT_vector_26199300505816
// MI455X (gfx1250) — matrix operand order repaired and checked by reading, not compiled, not run on hardware
//
#include <hip/hip_runtime.h>

#define DEPTH  6
#define HEADS  12
#define DH     64
#define DMODEL 768
#define MLPD   3072
#define SEQ    1024
#define BATCH  2
#define ROWS   (BATCH*SEQ)        // 2048
#define PROWS  (BATCH*(SEQ-1))    // 2046

typedef __attribute__((ext_vector_type(16))) __bf16 v16bf;
typedef __attribute__((ext_vector_type(8)))  float  v8f;
typedef __attribute__((address_space(3)))    __bf16 lds_bf16;

__device__ __forceinline__ __bf16 f2bf(float f) {
  union { float f; unsigned u; } v; v.f = f;
  unsigned r = v.u + 0x7FFFu + ((v.u >> 16) & 1u);   // round-to-nearest-even
  union { unsigned short s; __bf16 b; } o; o.s = (unsigned short)(r >> 16);
  return o.b;
}

// Build a 16-element bf16 A/B fragment from two contiguous 16-byte groups.
__device__ __forceinline__ v16bf ldfrag(const __bf16* p0, const __bf16* p1) {
  union U { uint4 u; __bf16 b[8]; } lo, hi;
  lo.u = *(const uint4*)p0;
  hi.u = *(const uint4*)p1;
  v16bf f;
#pragma unroll
  for (int i = 0; i < 8; ++i) { f[i] = lo.b[i]; f[i + 8] = hi.b[i]; }
  return f;
}

__device__ __forceinline__ v8f wmma_bf16(v16bf a, v16bf b, v8f c) {
  // D(16x16 f32) = A(16x32 bf16) x B(32x16 bf16) + C
  return __builtin_amdgcn_wmma_f32_16x16x32_bf16(false, a, false, b, (short)0, c,
                                                 false, false);
}

// CDNA5 async global->LDS copy (ASYNCcnt-tracked, no VGPR data movement).
__device__ __forceinline__ void async_copy_b128(unsigned lds_off, const void* gptr) {
  asm volatile("global_load_async_to_lds_b128 %0, %1, off"
               :: "v"(lds_off), "v"((unsigned long long)gptr)
               : "memory");
}
__device__ __forceinline__ void wait_async0() {
  asm volatile("s_wait_asynccnt 0" ::: "memory");
}

// ---------------------------------------------------------------- convert f32->bf16
__global__ void cvt_kernel(const float* __restrict__ in, __bf16* __restrict__ out, int n) {
  int i = blockIdx.x * blockDim.x + threadIdx.x;
  if (i < n) out[i] = f2bf(in[i]);
}

// ---------------------------------------------------------------- LayerNorm -> bf16
__global__ __launch_bounds__(256)
void ln_bf16_kernel(const float* __restrict__ x, const float* __restrict__ g,
                    const float* __restrict__ b, __bf16* __restrict__ out, int D) {
  const int row = blockIdx.x, tid = threadIdx.x;
  const float* xr = x + (size_t)row * D;
  __shared__ float red[256];
  float s = 0.f;
  for (int j = tid; j < D; j += 256) s += xr[j];
  red[tid] = s; __syncthreads();
  for (int o = 128; o > 0; o >>= 1) { if (tid < o) red[tid] += red[tid + o]; __syncthreads(); }
  float mean = red[0] / D;
  __syncthreads();
  float v = 0.f;
  for (int j = tid; j < D; j += 256) { float d = xr[j] - mean; v += d * d; }
  red[tid] = v; __syncthreads();
  for (int o = 128; o > 0; o >>= 1) { if (tid < o) red[tid] += red[tid + o]; __syncthreads(); }
  float rstd = rsqrtf(red[0] / D + 1e-5f);
  __bf16* orow = out + (size_t)row * D;
  for (int j = tid; j < D; j += 256) orow[j] = f2bf((xr[j] - mean) * rstd * g[j] + b[j]);
}

// LN of patch output, scattered to concat position (row -> b*1024 + 1 + i), f32 out.
__global__ __launch_bounds__(256)
void ln_scatter_kernel(const float* __restrict__ x, const float* __restrict__ g,
                       const float* __restrict__ b, float* __restrict__ out, int D) {
  const int row = blockIdx.x, tid = threadIdx.x;
  const float* xr = x + (size_t)row * D;
  __shared__ float red[256];
  float s = 0.f;
  for (int j = tid; j < D; j += 256) s += xr[j];
  red[tid] = s; __syncthreads();
  for (int o = 128; o > 0; o >>= 1) { if (tid < o) red[tid] += red[tid + o]; __syncthreads(); }
  float mean = red[0] / D;
  __syncthreads();
  float v = 0.f;
  for (int j = tid; j < D; j += 256) { float d = xr[j] - mean; v += d * d; }
  red[tid] = v; __syncthreads();
  for (int o = 128; o > 0; o >>= 1) { if (tid < o) red[tid] += red[tid + o]; __syncthreads(); }
  float rstd = rsqrtf(red[0] / D + 1e-5f);
  int orow = (row / (SEQ - 1)) * SEQ + (row % (SEQ - 1)) + 1;
  float* op = out + (size_t)orow * D;
  for (int j = tid; j < D; j += 256) op[j] = (xr[j] - mean) * rstd * g[j] + b[j];
}

// ---------------------------------------------------------------- cls token fill
__global__ void cls_fill_kernel(const float* __restrict__ cls, float* __restrict__ xtmp) {
  int i = blockIdx.x * blockDim.x + threadIdx.x;
  if (i < BATCH * DMODEL) {
    int bi = i / DMODEL, c = i % DMODEL;
    xtmp[(size_t)bi * SEQ * DMODEL + c] = cls[c];
  }
}

// ---------------------------------------------------------------- PEG depthwise conv + add
__global__ void peg_kernel(const float* __restrict__ xt, const float* __restrict__ w,
                           const float* __restrict__ pb, float* __restrict__ xo) {
  int i = blockIdx.x * blockDim.x + threadIdx.x;
  if (i >= BATCH * SEQ * DMODEL) return;
  int ch = i % DMODEL;
  int p  = (i / DMODEL) % SEQ;
  int bi = i / (DMODEL * SEQ);
  const float* base = xt + (size_t)bi * SEQ * DMODEL;
  float mid = base[(size_t)p * DMODEL + ch];
  float lf = (p > 0)       ? base[(size_t)(p - 1) * DMODEL + ch] : 0.f;
  float rt = (p < SEQ - 1) ? base[(size_t)(p + 1) * DMODEL + ch] : 0.f;
  xo[i] = mid + w[ch * 3 + 0] * lf + w[ch * 3 + 1] * mid + w[ch * 3 + 2] * rt + pb[ch];
}

// ---------------------------------------------------------------- generic bf16 WMMA GEMM
// C = act(A[MxK] * B[KxN] + bias) + resid, f32 and/or bf16 out (template-selected).
// Block tile 64x128, BK=32, 8 waves (each 32x32 via 4 WMMA accumulators).
// A tile staged via CDNA5 async global->LDS b128; B tile staged transposed [n][k].
template<bool BIAS, bool RES, bool ACT, bool OUTF, bool OUTB>
__global__ __launch_bounds__(256)
void gemm_bf16_kernel(const __bf16* __restrict__ A, const __bf16* __restrict__ B,
                      const float* __restrict__ bias, const float* __restrict__ resid,
                      float* __restrict__ Cf, __bf16* __restrict__ Cb,
                      int M, int N, int K) {
  __shared__ __align__(16) __bf16 As[64 * 32];   // [m][k]
  __shared__ __align__(16) __bf16 Bs[128 * 32];  // transposed: [n][k]
  const int tid = threadIdx.x;
  const int lane = tid & 31, wid = tid >> 5;
  const int half = lane >> 4, ln16 = lane & 15;
  const int m0 = blockIdx.x * 64;
  const int n0 = blockIdx.y * 128;
  const int wm = (wid >> 2) * 32;
  const int wn = (wid & 3) * 32;

  // A staging: one async b128 per thread (8 bf16 along k). Clamp out-of-range rows
  // (their products land in C rows the epilogue discards).
  const int arow = tid >> 2;
  const int akc  = (tid & 3) * 8;
  int ar = m0 + arow; if (ar > M - 1) ar = M - 1;
  const __bf16* agp = A + (size_t)ar * K + akc;
  const unsigned as_lds =
      (unsigned)(unsigned long long)(lds_bf16*)&As[arow * 32 + akc];

  // B staging: two b128 global reads (16 bf16 along n), transpose on LDS store.
  const int bk = tid >> 3;         // 0..31
  const int bn = (tid & 7) * 16;   // 0..112

  v8f acc[2][2];
#pragma unroll
  for (int mi = 0; mi < 2; ++mi)
#pragma unroll
    for (int ni = 0; ni < 2; ++ni)
#pragma unroll
      for (int r = 0; r < 8; ++r) acc[mi][ni][r] = 0.f;

  for (int k0 = 0; k0 < K; k0 += 32) {
    async_copy_b128(as_lds, (const void*)(agp + k0));

    {
      union { uint4 u; __bf16 b[8]; } t0, t1;
      const __bf16* bg = B + (size_t)(k0 + bk) * N + n0 + bn;
      t0.u = *(const uint4*)bg;
      t1.u = *(const uint4*)(bg + 8);
      if (k0 + 32 < K)
        __builtin_prefetch((const void*)(bg + (size_t)32 * N), 0, 3);
#pragma unroll
      for (int i = 0; i < 8; ++i) {
        Bs[(bn + i)     * 32 + bk] = t0.b[i];
        Bs[(bn + 8 + i) * 32 + bk] = t1.b[i];
      }
    }
    wait_async0();
    __syncthreads();

#pragma unroll
    for (int mi = 0; mi < 2; ++mi) {
      int rl = wm + mi * 16 + ln16;
      v16bf a = ldfrag(&As[rl * 32 + half * 16], &As[rl * 32 + half * 16 + 8]);
#pragma unroll
      for (int ni = 0; ni < 2; ++ni) {
        int nl = wn + ni * 16 + ln16;
        v16bf b = ldfrag(&Bs[nl * 32 + half * 16], &Bs[nl * 32 + half * 16 + 8]);
        acc[mi][ni] = wmma_bf16(a, b, acc[mi][ni]);
      }
    }
    __syncthreads();
  }

#pragma unroll
  for (int mi = 0; mi < 2; ++mi)
#pragma unroll
    for (int ni = 0; ni < 2; ++ni)
#pragma unroll
      for (int r = 0; r < 8; ++r) {
        int row = m0 + wm + mi * 16 + r + half * 8;
        int col = n0 + wn + ni * 16 + ln16;
        if (row < M) {
          float v = acc[mi][ni][r];
          if (BIAS) v += bias[col];
          if (ACT)  v = 0.5f * v * (1.f + erff(v * 0.70710678118654752f));
          if (RES)  v += resid[(size_t)row * N + col];
          if (OUTF) Cf[(size_t)row * N + col] = v;
          if (OUTB) Cb[(size_t)row * N + col] = f2bf(v);
        }
      }
}

// ---------------------------------------------------------------- RoPE + split/convert
// qkv f32 [2048][3*768] -> qh,kh bf16 [b][h][n][64], vt bf16 [b][h][64][n]
__global__ void rope_split_kernel(const float* __restrict__ qkv,
                                  const float* __restrict__ inv_freq,
                                  __bf16* __restrict__ qh, __bf16* __restrict__ kh,
                                  __bf16* __restrict__ vt) {
  int i = blockIdx.x * blockDim.x + threadIdx.x;
  if (i >= BATCH * SEQ * HEADS * 32) return;
  int d2 = i & 31;
  int h  = (i >> 5) % HEADS;
  int n  = (i / (32 * HEADS)) % SEQ;
  int bi = i / (32 * HEADS * SEQ);
  size_t row = (size_t)(bi * SEQ + n);
  const float* base = qkv + row * (3 * HEADS * DH) + h * DH;
  int d = 2 * d2;
  float q0 = base[d],        q1 = base[d + 1];
  float k0 = base[768 + d],  k1 = base[768 + d + 1];
  float v0 = base[1536 + d], v1 = base[1536 + d + 1];
  if (d2 < 16) {  // rotary on first 32 dims
    float fr = (float)n * inv_freq[d2];
    float cs = cosf(fr), sn = sinf(fr);
    float tq0 = q0 * cs - q1 * sn, tq1 = q1 * cs + q0 * sn;
    float tk0 = k0 * cs - k1 * sn, tk1 = k1 * cs + k0 * sn;
    q0 = tq0; q1 = tq1; k0 = tk0; k1 = tk1;
  }
  size_t bh = (size_t)(bi * HEADS + h);
  size_t qo = (bh * SEQ + n) * DH + d;
  qh[qo] = f2bf(q0); qh[qo + 1] = f2bf(q1);
  kh[qo] = f2bf(k0); kh[qo + 1] = f2bf(k1);
  vt[(bh * DH + d) * SEQ + n]     = f2bf(v0);
  vt[(bh * DH + d + 1) * SEQ + n] = f2bf(v1);
}

// ---------------------------------------------------------------- flash attention
// 4 waves / block, each wave owns a 16-row query tile; keys streamed in chunks of 32.
__global__ __launch_bounds__(128)
void attn_kernel(const __bf16* __restrict__ qh, const __bf16* __restrict__ kh,
                 const __bf16* __restrict__ vt, const float* __restrict__ mask,
                 __bf16* __restrict__ outp) {
  const int lane = threadIdx.x & 31, wid = threadIdx.x >> 5;
  const int half = lane >> 4, ln16 = lane & 15;
  const int h = blockIdx.y, bi = blockIdx.z;
  const size_t bh = (size_t)(bi * HEADS + h);
  const int qbase = blockIdx.x * 64 + wid * 16;
  const __bf16* qp = qh + bh * SEQ * DH;
  const __bf16* kp = kh + bh * SEQ * DH;
  const __bf16* vp = vt + bh * DH * SEQ;
  const float*  mp = mask + bh * (size_t)SEQ * SEQ;
  __shared__ __align__(16) __bf16 pst[4][16 * 32];  // per-wave P staging

  v16bf aQ[2];
  {
    const __bf16* q0 = qp + (size_t)(qbase + ln16) * DH;
    aQ[0] = ldfrag(q0 + half * 16,      q0 + half * 16 + 8);
    aQ[1] = ldfrag(q0 + 32 + half * 16, q0 + 32 + half * 16 + 8);
  }
  v8f oacc[4];
#pragma unroll
  for (int nt = 0; nt < 4; ++nt)
#pragma unroll
    for (int r = 0; r < 8; ++r) oacc[nt][r] = 0.f;
  float mrun[8], lrun[8];
#pragma unroll
  for (int r = 0; r < 8; ++r) { mrun[r] = -1e30f; lrun[r] = 0.f; }
  const float scale = 0.125f;  // 64^-0.5

  for (int c = 0; c < SEQ / 32; ++c) {
    const int cb = c * 32;
    v8f s[2];
#pragma unroll
    for (int t = 0; t < 2; ++t) {  // S = Q(16x64) * K^T, two 16-col tiles
      const __bf16* kb = kp + (size_t)(cb + t * 16 + ln16) * DH;
      v16bf b0 = ldfrag(kb + half * 16,      kb + half * 16 + 8);
      v16bf b1 = ldfrag(kb + 32 + half * 16, kb + 32 + half * 16 + 8);
      v8f z;
#pragma unroll
      for (int r = 0; r < 8; ++r) z[r] = 0.f;
      z    = wmma_bf16(aQ[0], b0, z);
      s[t] = wmma_bf16(aQ[1], b1, z);
    }
    // online softmax in C-fragment layout (row = r + 8*half, col = ln16)
#pragma unroll
    for (int r = 0; r < 8; ++r) {
      int row = qbase + r + half * 8;
      int c0  = cb + ln16;
      float sv0 = s[0][r] * scale + (1.f - mp[(size_t)row * SEQ + c0])      * -1e9f;
      float sv1 = s[1][r] * scale + (1.f - mp[(size_t)row * SEQ + c0 + 16]) * -1e9f;
      float mx = fmaxf(sv0, sv1);
#pragma unroll
      for (int m = 1; m < 16; m <<= 1) mx = fmaxf(mx, __shfl_xor(mx, m, 32));
      float mn  = fmaxf(mrun[r], mx);
      float fac = __expf(mrun[r] - mn);
      mrun[r] = mn;
      float p0 = __expf(sv0 - mn), p1 = __expf(sv1 - mn);
      float rs = p0 + p1;
#pragma unroll
      for (int m = 1; m < 16; m <<= 1) rs += __shfl_xor(rs, m, 32);
      lrun[r] = lrun[r] * fac + rs;
#pragma unroll
      for (int nt = 0; nt < 4; ++nt) oacc[nt][r] *= fac;
      int ml = r + half * 8;
      pst[wid][ml * 32 + ln16]      = f2bf(p0);
      pst[wid][ml * 32 + 16 + ln16] = f2bf(p1);
    }
    __syncthreads();
    // P (16x32) as A-fragment; V chunk (32x64) as four B-fragments from vt[dh][n]
    v16bf aP = ldfrag(&pst[wid][ln16 * 32 + half * 16],
                      &pst[wid][ln16 * 32 + half * 16 + 8]);
#pragma unroll
    for (int nt = 0; nt < 4; ++nt) {
      const __bf16* vb = vp + (size_t)(nt * 16 + ln16) * SEQ + cb + half * 16;
      v16bf bV = ldfrag(vb, vb + 8);
      oacc[nt] = wmma_bf16(aP, bV, oacc[nt]);
    }
    __syncthreads();
  }
#pragma unroll
  for (int nt = 0; nt < 4; ++nt)
#pragma unroll
    for (int r = 0; r < 8; ++r) {
      int row = qbase + r + half * 8;
      outp[(size_t)(bi * SEQ + row) * (HEADS * DH) + h * DH + nt * 16 + ln16] =
          f2bf(oacc[nt][r] / lrun[r]);
    }
}

// ---------------------------------------------------------------- host side
extern "C" void kernel_launch(void* const* d_in, const int* in_sizes, int n_in,
                              void* d_out, int out_size, void* d_ws, size_t ws_size,
                              hipStream_t stream) {
  const float* filaments = (const float*)d_in[0];
  const float* mask      = (const float*)d_in[1];
  const float* ln_p1_g   = (const float*)d_in[2];
  const float* ln_p1_b   = (const float*)d_in[3];
  const float* W_patch   = (const float*)d_in[4];
  const float* b_patch   = (const float*)d_in[5];
  const float* ln_p2_g   = (const float*)d_in[6];
  const float* ln_p2_b   = (const float*)d_in[7];
  const float* peg_w     = (const float*)d_in[8];
  const float* peg_b     = (const float*)d_in[9];
  const float* cls_token = (const float*)d_in[10];
  const float* ln_a_g    = (const float*)d_in[11];
  const float* ln_a_b    = (const float*)d_in[12];
  const float* W_qkv     = (const float*)d_in[13];
  const float* W_o       = (const float*)d_in[14];
  const float* b_o       = (const float*)d_in[15];
  const float* inv_freq  = (const float*)d_in[16];
  const float* ln_f_g    = (const float*)d_in[17];
  const float* ln_f_b    = (const float*)d_in[18];
  const float* W1        = (const float*)d_in[19];
  const float* b1        = (const float*)d_in[20];
  const float* W2        = (const float*)d_in[21];
  const float* b2        = (const float*)d_in[22];
  (void)in_sizes; (void)n_in; (void)out_size; (void)ws_size;

  char* ws = (char*)d_ws;
  size_t off = 0;
  auto alloc = [&](size_t bytes) -> void* {
    off = (off + 255) & ~(size_t)255;
    void* p = ws + off;
    off += bytes;
    return p;
  };

  float*  xf32   = (float*) alloc((size_t)ROWS * DMODEL * 4);
  float*  xtmp   = (float*) alloc((size_t)ROWS * DMODEL * 4);
  __bf16* hbuf   = (__bf16*)alloc((size_t)ROWS * DMODEL * 2);
  float*  qkvbuf = (float*) alloc((size_t)ROWS * 3 * DMODEL * 4);  // also patch GEMM out
  __bf16* qhb    = (__bf16*)alloc((size_t)BATCH * HEADS * SEQ * DH * 2);
  __bf16* khb    = (__bf16*)alloc((size_t)BATCH * HEADS * SEQ * DH * 2);
  __bf16* vtb    = (__bf16*)alloc((size_t)BATCH * HEADS * SEQ * DH * 2);
  __bf16* attno  = (__bf16*)alloc((size_t)ROWS * DMODEL * 2);
  __bf16* midb   = (__bf16*)alloc((size_t)ROWS * MLPD * 2);
  __bf16* Wp     = (__bf16*)alloc((size_t)DMODEL * DMODEL * 2);
  __bf16* Wq     = (__bf16*)alloc((size_t)DMODEL * 3 * DMODEL * 2);
  __bf16* Wo     = (__bf16*)alloc((size_t)DMODEL * DMODEL * 2);
  __bf16* Wt     = (__bf16*)alloc((size_t)DMODEL * MLPD * 2);     // per-layer W1/W2 staging

  auto cvt = [&](const float* src, __bf16* dst, int n) {
    cvt_kernel<<<(n + 255) / 256, 256, 0, stream>>>(src, dst, n);
  };

  cvt(W_patch, Wp, DMODEL * DMODEL);
  cvt(W_qkv,   Wq, DMODEL * 3 * DMODEL);
  cvt(W_o,     Wo, DMODEL * DMODEL);

  // patch embed: LN(filaments) @ W_patch + b_patch
  ln_bf16_kernel<<<PROWS, 256, 0, stream>>>(filaments, ln_p1_g, ln_p1_b, hbuf, DMODEL);
  gemm_bf16_kernel<true, false, false, true, false>
      <<<dim3((PROWS + 63) / 64, DMODEL / 128), 256, 0, stream>>>(
      hbuf, Wp, b_patch, nullptr, qkvbuf, nullptr, PROWS, DMODEL, DMODEL);

  // LN2 -> concat positions; cls row; PEG conv + add
  ln_scatter_kernel<<<PROWS, 256, 0, stream>>>(qkvbuf, ln_p2_g, ln_p2_b, xtmp, DMODEL);
  cls_fill_kernel<<<(BATCH * DMODEL + 255) / 256, 256, 0, stream>>>(cls_token, xtmp);
  peg_kernel<<<(BATCH * SEQ * DMODEL + 255) / 256, 256, 0, stream>>>(xtmp, peg_w, peg_b, xf32);

  for (int i = 0; i < DEPTH; ++i) {
    ln_bf16_kernel<<<ROWS, 256, 0, stream>>>(xf32, ln_a_g + i * DMODEL, ln_a_b + i * DMODEL,
                                             hbuf, DMODEL);
    gemm_bf16_kernel<false, false, false, true, false>
        <<<dim3(ROWS / 64, (3 * DMODEL) / 128), 256, 0, stream>>>(
        hbuf, Wq, nullptr, nullptr, qkvbuf, nullptr, ROWS, 3 * DMODEL, DMODEL);
    rope_split_kernel<<<(BATCH * SEQ * HEADS * 32 + 255) / 256, 256, 0, stream>>>(
        qkvbuf, inv_freq, qhb, khb, vtb);
    attn_kernel<<<dim3(SEQ / 64, HEADS, BATCH), 128, 0, stream>>>(qhb, khb, vtb, mask, attno);
    gemm_bf16_kernel<true, true, false, true, false>
        <<<dim3(ROWS / 64, DMODEL / 128), 256, 0, stream>>>(
        attno, Wo, b_o, xf32, xf32, nullptr, ROWS, DMODEL, DMODEL);

    ln_bf16_kernel<<<ROWS, 256, 0, stream>>>(xf32, ln_f_g + i * DMODEL, ln_f_b + i * DMODEL,
                                             hbuf, DMODEL);
    cvt(W1 + (size_t)i * DMODEL * MLPD, Wt, DMODEL * MLPD);
    gemm_bf16_kernel<true, false, true, false, true>
        <<<dim3(ROWS / 64, MLPD / 128), 256, 0, stream>>>(
        hbuf, Wt, b1 + i * MLPD, nullptr, nullptr, midb, ROWS, MLPD, DMODEL);
    cvt(W2 + (size_t)i * MLPD * DMODEL, Wt, MLPD * DMODEL);
    float* xout = (i == DEPTH - 1) ? (float*)d_out : xf32;
    gemm_bf16_kernel<true, true, false, true, false>
        <<<dim3(ROWS / 64, DMODEL / 128), 256, 0, stream>>>(
        midb, Wt, b2 + i * DMODEL, xf32, xout, nullptr, ROWS, DMODEL, MLPD);
  }
}
